// Interactor_57887569215866
// MI455X (gfx1250) — compile-verified, matrix operand order checked by reading, not run on hardware
//
#include <hip/hip_runtime.h>
#include <hip/hip_bf16.h>
#include <math.h>

// ---------------------------------------------------------------------------
// Types for CDNA5 WMMA (wave32): D(16x16 f32) = A(16x32 f16) x B(32x16 f16) + C
// ---------------------------------------------------------------------------
typedef __attribute__((ext_vector_type(16))) _Float16 v16h;
typedef __attribute__((ext_vector_type(8)))  _Float16 v8h;
typedef __attribute__((ext_vector_type(8)))  float    v8f;

#define NOBJ   8
#define HDIM   256
#define NPAIR  64
#define NSCENE 2048
#define NTOK   16384

// ---------------------------------------------------------------------------
// Workspace layout (bytes, 256-aligned). Packed weights are f16 in WMMA-B
// fragment order: tile (nt,kt) -> 32 lanes x 16 halves contiguous (1KB/tile).
// ---------------------------------------------------------------------------
constexpr size_t aup(size_t x) { return (x + 255) & ~size_t(255); }
constexpr size_t PW_SMALL = (size_t)16 * 1 * 512 * 2;  // ntN=16, ktN=1 (K pad 32)
constexpr size_t PW_BIG   = (size_t)16 * 8 * 512 * 2;  // ntN=16, ktN=8 (K=256)
constexpr size_t PW_ATT   = (size_t)7  * 8 * 512 * 2;  // ntN=7  (N pad 112)

constexpr size_t OFF_PWE   = 0;
constexpr size_t OFF_PWI   = aup(OFF_PWE  + PW_SMALL);
constexpr size_t OFF_PWA   = aup(OFF_PWI  + PW_SMALL);
constexpr size_t OFF_PWEC  = aup(OFF_PWA  + PW_SMALL);
constexpr size_t OFF_PWIC  = aup(OFF_PWEC + PW_SMALL);
constexpr size_t OFF_PW2   = aup(OFF_PWIC + PW_SMALL);
constexpr size_t OFF_PW2C  = aup(OFF_PW2  + PW_BIG);
constexpr size_t OFF_PWAT  = aup(OFF_PW2C + PW_BIG);
constexpr size_t OFF_WCM   = aup(OFF_PWAT + PW_ATT);          // 257 f32
constexpr size_t OFF_WCC   = aup(OFF_WCM + 257 * 4);          // 257 f32
constexpr size_t OFF_S     = aup(OFF_WCC + 257 * 4);          // 2048*64 f32
constexpr size_t OFF_SC    = aup(OFF_S   + (size_t)NSCENE * 64 * 4);
constexpr size_t OFF_PM    = aup(OFF_SC  + (size_t)NSCENE * 64 * 4);
constexpr size_t OFF_PART  = aup(OFF_PM  + (size_t)NSCENE * 64 * 4);  // 2048*4 f32
constexpr size_t OFF_STATS = aup(OFF_PART + (size_t)NSCENE * 4 * 4);  // 6 f32
constexpr size_t OFF_E     = aup(OFF_STATS + 256);            // 2048*8*256 f32
constexpr size_t OFF_EC    = aup(OFF_E + (size_t)NSCENE * 2048 * 4);
// total ~35.6 MB

// ---------------------------------------------------------------------------
// Kernel 0: pack fp32 weight W (Nin x Kin, row-major, out-major) into f16
// WMMA-B fragment order with zero padding.
//   lane = (n%16) + 16*h ; element e: k = kt*32 + (e<8 ? 8h+e : 16+8h+(e-8))
// ---------------------------------------------------------------------------
__global__ void k_pack(const float* __restrict__ W, _Float16* __restrict__ dst,
                       int ktN, int Nin, int Kin, int total) {
  int idx = blockIdx.x * blockDim.x + threadIdx.x;
  if (idx >= total) return;
  int e    = idx & 15;
  int lane = (idx >> 4) & 31;
  int tile = idx >> 9;
  int kt = tile % ktN;
  int nt = tile / ktN;
  int n = nt * 16 + (lane & 15);
  int h = lane >> 4;
  int koff = (e < 8) ? (8 * h + e) : (16 + 8 * h + (e - 8));
  int k = kt * 32 + koff;
  float v = (n < Nin && k < Kin) ? W[(size_t)n * Kin + k] : 0.0f;
  dst[(size_t)tile * 512 + lane * 16 + e] = (_Float16)v;
}

// ---------------------------------------------------------------------------
// Kernel 0b: collapse s-path double linear: out[k] = sum_h Ws2[h]*Ws1[h,k],
// out[256] = Ws2 . bs1 + bs2   (no nonlinearity between the two linears)
// ---------------------------------------------------------------------------
__global__ void k_comb(const float* __restrict__ Ws1, const float* __restrict__ bs1,
                       const float* __restrict__ Ws2, const float* __restrict__ bs2,
                       float* __restrict__ out) {
  int k = threadIdx.x;
  float acc = 0.0f;
  for (int h = 0; h < 256; ++h) acc += Ws2[h] * Ws1[h * 256 + k];
  out[k] = acc;
  if (k == 0) {
    float b = bs2[0];
    for (int h = 0; h < 256; ++h) b += Ws2[h] * bs1[h];
    out[256] = b;
  }
}

// ---------------------------------------------------------------------------
// WMMA helpers
// ---------------------------------------------------------------------------
static __device__ __forceinline__ v16h
load_a_frag(const _Float16* __restrict__ A, int astride, int m, int h, int k0) {
  const _Float16* p = A + (size_t)m * astride + k0 + 8 * h;
  v8h lo = *(const v8h*)p;          // K = k0 + 8h + [0..7]
  v8h hi = *(const v8h*)(p + 16);   // K = k0 + 16 + 8h + [0..7]
  return __builtin_shufflevector(lo, hi, 0, 1, 2, 3, 4, 5, 6, 7,
                                 8, 9, 10, 11, 12, 13, 14, 15);
}

static __device__ __forceinline__ v8f wmma32(v16h a, v16h b, v8f c) {
  return __builtin_amdgcn_wmma_f32_16x16x32_f16(false, a, false, b,
                                                (short)0, c, false, false);
}

// relu(A(64x32) @ W^T + b) -> Out (64x256 f16 LDS). Each wave owns two
// N-strips (nt = w, w+8); B fragment loaded ONCE and reused across 4 row tiles.
static __device__ __forceinline__ void
gemm_relu_store(const _Float16* __restrict__ Alds, const _Float16* __restrict__ pw,
                const float* __restrict__ bias, _Float16* __restrict__ Out,
                int lane, int w) {
  const int h  = lane >> 4;
  const int nl = lane & 15;
  for (int q = 0; q < 2; ++q) {
    const int nt = w + q * 8;
    const int n  = nt * 16 + nl;
    const v16h b = *(const v16h*)(pw + (size_t)nt * 512 + lane * 16);
    const float bn = bias[n];
    for (int mt = 0; mt < 4; ++mt) {
      const int m0 = mt * 16;
      v16h a = load_a_frag(Alds, 32, m0 + nl, h, 0);
      v8f c = {};
      c = wmma32(a, b, c);
#pragma unroll
      for (int r = 0; r < 8; ++r) {
        float v = fmaxf(c[r] + bn, 0.0f);
        Out[(size_t)(m0 + r + 8 * h) * 256 + n] = (_Float16)v;
      }
    }
  }
}

// (Buf(64x256) @ W^T + b) with per-row weights, summed over i (row = i*8+j):
// out[j][n] = sum_i (T2[i*8+j][n] + b[n]) * weight(i,j).
// B fragments (8 k-steps) preloaded into registers per N-strip.
// use_pm=true: weight = pmArr[i*8+j], bias factor pmsum[j]
// use_pm=false: weight = (i!=j), bias factor 7
static __device__ __forceinline__ void
gemm_wsum_store(const _Float16* __restrict__ Buf, const _Float16* __restrict__ pw,
                const float* __restrict__ bias, const float* __restrict__ pmArr,
                const float* __restrict__ pmsum, bool use_pm,
                float* __restrict__ dst, int lane, int w) {
  const int h  = lane >> 4;
  const int nl = lane & 15;
  for (int q = 0; q < 2; ++q) {
    const int nt = w + q * 8;
    const int n  = nt * 16 + nl;
    v16h bfr[8];
#pragma unroll
    for (int kt = 0; kt < 8; ++kt)
      bfr[kt] = *(const v16h*)(pw + (size_t)(nt * 8 + kt) * 512 + lane * 16);
    v8f acc = {};
    for (int mt = 0; mt < 4; ++mt) {
      v8f c = {};
#pragma unroll
      for (int kt = 0; kt < 8; ++kt) {
        v16h a = load_a_frag(Buf, 256, mt * 16 + nl, h, kt * 32);
        c = wmma32(a, bfr[kt], c);
      }
      const int i = 2 * mt + h;  // row = mt*16 + r + 8h -> i = 2mt+h, j = r
#pragma unroll
      for (int r = 0; r < 8; ++r) {
        float wgt = use_pm ? pmArr[i * 8 + r] : ((i == r) ? 0.0f : 1.0f);
        acc[r] += c[r] * wgt;
      }
    }
#pragma unroll
    for (int r = 0; r < 8; ++r) acc[r] += __shfl_xor(acc[r], 16, 32);
    if (lane < 16) {
      const float bn = bias[n];
#pragma unroll
      for (int r = 0; r < 8; ++r) {
        float bf = use_pm ? pmsum[r] : 7.0f;
        dst[(size_t)r * 256 + n] = acc[r] + bn * bf;
      }
    }
  }
}

// ---------------------------------------------------------------------------
// Kernel 1: one workgroup (8 wave32) per scene. Fully fused pair pipeline.
// ---------------------------------------------------------------------------
__global__ __launch_bounds__(256) void k_scene(
    const float* __restrict__ zdyn, const float* __restrict__ zctx,
    const float* __restrict__ zmc,
    const float* __restrict__ b_e, const float* __restrict__ b_i,
    const float* __restrict__ b_a, const float* __restrict__ b_eC,
    const float* __restrict__ b_iC, const float* __restrict__ b_2,
    const float* __restrict__ b_2C, const float* __restrict__ b_at,
    const float* __restrict__ W_at2, const float* __restrict__ b_at2,
    const _Float16* __restrict__ pwE, const _Float16* __restrict__ pwI,
    const _Float16* __restrict__ pwA, const _Float16* __restrict__ pwEC,
    const _Float16* __restrict__ pwIC, const _Float16* __restrict__ pw2,
    const _Float16* __restrict__ pw2C, const _Float16* __restrict__ pwAT,
    const float* __restrict__ wcM, const float* __restrict__ wcC,
    float* __restrict__ s_ws, float* __restrict__ sC_ws,
    float* __restrict__ pm_ws, float* __restrict__ part_ws,
    float* __restrict__ E_ws, float* __restrict__ EC_ws) {
  __shared__ __attribute__((aligned(16))) float    objf[8 * 16];
  __shared__ __attribute__((aligned(16))) _Float16 xf[64 * 32];
  __shared__ __attribute__((aligned(16))) _Float16 ff[64 * 32];
  __shared__ __attribute__((aligned(16))) _Float16 fcf[64 * 32];
  __shared__ __attribute__((aligned(16))) _Float16 Buf[64 * 256];
  __shared__ __attribute__((aligned(16))) _Float16 attb[64 * 112];
  __shared__ float pmArr[64], sArr[64], sCArr[64], pmsum[8];

  const int scene = blockIdx.x;
  const int tid   = threadIdx.x;
  const int lane  = tid & 31;
  const int w     = tid >> 5;

  // Warm L2/L0 for this wave's W_at strip while features are being built.
  if (w < 7) {
#pragma unroll
    for (int kt = 0; kt < 8; ++kt)
      __builtin_prefetch(pwAT + (size_t)(w * 8 + kt) * 512 + lane * 16, 0, 3);
  }

  // ---- Phase A: load per-object features, build pair rows (f16) -----------
  if (tid < 128) {
    int o = tid >> 4, f = tid & 15;
    int t = scene * NOBJ + o;
    float v;
    if (f < 2)        v = zdyn[t * 2 + f];         // v (2)
    else if (f < 14)  v = zctx[t * 12 + (f - 2)];  // ct (12)
    else if (f == 14) v = zmc[t * 2 + 0];          // zm
    else              v = zmc[t * 2 + 1];          // zc
    objf[o * 16 + f] = v;
  }
  __syncthreads();
  if (tid < 64) {
    int i = tid >> 3, j = tid & 7, r = tid;
    const float* cj = &objf[j * 16];
    const float* ci = &objf[i * 16];
    _Float16* X = &xf[r * 32];
    _Float16* F = &ff[r * 32];
    _Float16* G = &fcf[r * 32];
    // x_12 = [cd_j(14), cd_i(14), 0 pad]
    for (int k = 0; k < 14; ++k) { X[k] = (_Float16)cj[k]; X[14 + k] = (_Float16)ci[k]; }
    X[28] = X[29] = X[30] = X[31] = (_Float16)0.0f;
    // f_12 = [zm_j, cd_j(14), zm_i, cd_i(14), 0 pad]
    F[0] = (_Float16)cj[14];
    for (int k = 0; k < 14; ++k) F[1 + k] = (_Float16)cj[k];
    F[15] = (_Float16)ci[14];
    for (int k = 0; k < 14; ++k) F[16 + k] = (_Float16)ci[k];
    F[30] = F[31] = (_Float16)0.0f;
    // fc_12 = [ct_j(12), ct_i(12), zc_j, zc_i, v_j(2), v_i(2), 0 pad]
    for (int k = 0; k < 12; ++k) { G[k] = (_Float16)cj[2 + k]; G[12 + k] = (_Float16)ci[2 + k]; }
    G[24] = (_Float16)cj[15]; G[25] = (_Float16)ci[15];
    G[26] = (_Float16)cj[0];  G[27] = (_Float16)cj[1];
    G[28] = (_Float16)ci[0];  G[29] = (_Float16)ci[1];
    G[30] = G[31] = (_Float16)0.0f;
  }
  __syncthreads();

  // ---- Phase B: A_emb = relu(x12 @ W_a^T + b_a) -> Buf --------------------
  gemm_relu_store(xf, pwA, b_a, Buf, lane, w);
  __syncthreads();

  // ---- Phase C: att = tanh(A_emb @ W_at^T + b_at) -> attb (64x112) --------
  if (w < 7) {
    const int h  = lane >> 4;
    const int nl = lane & 15;
    const int nt = w;
    const int n  = nt * 16 + nl;
    v16h bfr[8];
#pragma unroll
    for (int kt = 0; kt < 8; ++kt)
      bfr[kt] = *(const v16h*)(pwAT + (size_t)(nt * 8 + kt) * 512 + lane * 16);
    const float bn = (n < 100) ? b_at[n] : 0.0f;
    for (int mt = 0; mt < 4; ++mt) {
      v8f c = {};
#pragma unroll
      for (int kt = 0; kt < 8; ++kt) {
        v16h a = load_a_frag(Buf, 256, mt * 16 + nl, h, kt * 32);
        c = wmma32(a, bfr[kt], c);
      }
#pragma unroll
      for (int r = 0; r < 8; ++r)
        attb[(size_t)(mt * 16 + r + 8 * h) * 112 + n] = (_Float16)tanhf(c[r] + bn);
    }
  }
  __syncthreads();

  // ---- Phase D: presents/pm ----------------------------------------------
  if (tid < 64) {
    int r = tid;
    float acc = b_at2[0];
    for (int k = 0; k < 100; ++k) acc += (float)attb[r * 112 + k] * W_at2[k];
    float p = 1.0f / (1.0f + expf(-acc));
    float m = ((r >> 3) == (r & 7)) ? 0.0f : p;  // kill diagonal i==j
    pmArr[r] = m;
    pm_ws[(size_t)scene * 64 + r] = m;
  }
  __syncthreads();
  if (tid < 8) {
    float s = 0.0f;
    for (int i = 0; i < 8; ++i) s += pmArr[i * 8 + tid];
    pmsum[tid] = s;
  }
  __syncthreads();

  // ---- Phase E: I_emb -> Buf ; s = I_emb . w_combM + bias -----------------
  gemm_relu_store(ff, pwI, b_i, Buf, lane, w);
  __syncthreads();
  if (tid < 64) {
    float acc = wcM[256];
    for (int n = 0; n < 256; ++n) acc += (float)Buf[tid * 256 + n] * wcM[n];
    sArr[tid] = acc;
    s_ws[(size_t)scene * 64 + tid] = acc;
  }
  __syncthreads();

  // ---- Phase F: I_embC -> Buf ; sC ---------------------------------------
  gemm_relu_store(fcf, pwIC, b_iC, Buf, lane, w);
  __syncthreads();
  if (tid < 64) {
    float acc = wcC[256];
    for (int n = 0; n < 256; ++n) acc += (float)Buf[tid * 256 + n] * wcC[n];
    sCArr[tid] = acc;
    sC_ws[(size_t)scene * 64 + tid] = acc;
  }
  // Prefetch this wave's W_2 / W_2C strips for phases G/H.
  {
#pragma unroll
    for (int kt = 0; kt < 8; ++kt) {
      __builtin_prefetch(pw2  + (size_t)(w * 8 + kt) * 512 + lane * 16, 0, 3);
      __builtin_prefetch(pw2  + (size_t)((w + 8) * 8 + kt) * 512 + lane * 16, 0, 3);
      __builtin_prefetch(pw2C + (size_t)(w * 8 + kt) * 512 + lane * 16, 0, 3);
      __builtin_prefetch(pw2C + (size_t)((w + 8) * 8 + kt) * 512 + lane * 16, 0, 3);
    }
  }
  __syncthreads();
  if (tid == 0) {  // per-scene BN partials (sum, sumsq) for M and C paths
    float s0 = 0, s1 = 0, s2 = 0, s3 = 0;
    for (int r = 0; r < 64; ++r) {
      s0 += sArr[r];  s1 += sArr[r] * sArr[r];
      s2 += sCArr[r]; s3 += sCArr[r] * sCArr[r];
    }
    part_ws[scene * 4 + 0] = s0; part_ws[scene * 4 + 1] = s1;
    part_ws[scene * 4 + 2] = s2; part_ws[scene * 4 + 3] = s3;
  }

  // ---- Phase G: E_emb -> Buf ; E[j] = sum_i (E_emb@W_2^T + b_2)*pm --------
  __syncthreads();
  gemm_relu_store(xf, pwE, b_e, Buf, lane, w);
  __syncthreads();
  gemm_wsum_store(Buf, pw2, b_2, pmArr, pmsum, true,
                  E_ws + (size_t)scene * 2048, lane, w);
  __syncthreads();

  // ---- Phase H: E_embC -> Buf ; E_C with off-diagonal mask ----------------
  gemm_relu_store(fcf, pwEC, b_eC, Buf, lane, w);
  __syncthreads();
  gemm_wsum_store(Buf, pw2C, b_2C, pmArr, pmsum, false,
                  EC_ws + (size_t)scene * 2048, lane, w);
}

// ---------------------------------------------------------------------------
// Kernel 2: global BN stats over 131072 elements (both paths)
// ---------------------------------------------------------------------------
__global__ void k_stats(const float* __restrict__ part,
                        const float* __restrict__ g_M, const float* __restrict__ be_M,
                        const float* __restrict__ g_C, const float* __restrict__ be_C,
                        float* __restrict__ stats) {
  __shared__ float red[256 * 4];
  int t = threadIdx.x;
  float a0 = 0, a1 = 0, a2 = 0, a3 = 0;
  for (int s = t; s < NSCENE; s += 256) {
    a0 += part[s * 4 + 0]; a1 += part[s * 4 + 1];
    a2 += part[s * 4 + 2]; a3 += part[s * 4 + 3];
  }
  red[t] = a0; red[256 + t] = a1; red[512 + t] = a2; red[768 + t] = a3;
  __syncthreads();
  for (int o = 128; o; o >>= 1) {
    if (t < o) {
      red[t] += red[t + o];             red[256 + t] += red[256 + t + o];
      red[512 + t] += red[512 + t + o]; red[768 + t] += red[768 + t + o];
    }
    __syncthreads();
  }
  if (t == 0) {
    const float N = (float)(NSCENE * 64);
    float mM = red[0] / N, vM = red[256] / N - mM * mM;
    float mC = red[512] / N, vC = red[768] / N - mC * mC;
    stats[0] = mM; stats[1] = g_M[0] / sqrtf(vM + 1e-5f); stats[2] = be_M[0];
    stats[3] = mC; stats[4] = g_C[0] / sqrtf(vC + 1e-5f); stats[5] = be_C[0];
  }
}

// ---------------------------------------------------------------------------
// Kernel 3: scale/scaleC, 256->2 projection, L2-normalize, final outputs
// ---------------------------------------------------------------------------
static __device__ __forceinline__ float softplusf(float x) {
  return (x > 20.0f) ? x : log1pf(expf(x));
}

__global__ __launch_bounds__(256) void k_final(
    const float* __restrict__ s_ws, const float* __restrict__ sC_ws,
    const float* __restrict__ pm_ws, const float* __restrict__ E_ws,
    const float* __restrict__ EC_ws, const float* __restrict__ stats,
    const float* __restrict__ W_d, const float* __restrict__ W_dC,
    float* __restrict__ out) {
  const int scene = blockIdx.x;
  const int j = threadIdx.x >> 5;  // one wave per object column j
  const int l = threadIdx.x & 31;
  const float meanM = stats[0], gsM = stats[1], beM = stats[2];
  const float meanC = stats[3], gsC = stats[4], beC = stats[5];

  float term = 0.0f, termC = 0.0f;
  if (l < 8) {
    size_t idx = (size_t)scene * 64 + l * 8 + j;  // row = i*8 + j, i = l
    float nm = (s_ws[idx] - meanM) * gsM + beM;
    term = softplusf(nm) * pm_ws[idx];
    if (l != j) {
      float nc = (sC_ws[idx] - meanC) * gsC + beC;
      termC = softplusf(nc);
    }
  }
  for (int o = 16; o; o >>= 1) {
    term  += __shfl_xor(term, o, 32);
    termC += __shfl_xor(termC, o, 32);
  }

  const float* E  = E_ws  + (size_t)scene * 2048 + j * 256;
  const float* EC = EC_ws + (size_t)scene * 2048 + j * 256;
  float d0 = 0, d1 = 0, e0 = 0, e1 = 0;
  for (int n = l; n < 256; n += 32) {
    float ev = E[n];  d0 += ev * W_d[n];  d1 += ev * W_d[256 + n];
    float ec = EC[n]; e0 += ec * W_dC[n]; e1 += ec * W_dC[256 + n];
  }
  for (int o = 16; o; o >>= 1) {
    d0 += __shfl_xor(d0, o, 32); d1 += __shfl_xor(d1, o, 32);
    e0 += __shfl_xor(e0, o, 32); e1 += __shfl_xor(e1, o, 32);
  }
  if (l == 0) {
    float nm  = sqrtf(d0 * d0 + d1 * d1);
    float inv = 1.0f / fmaxf(nm, 1e-12f);
    float m0 = term * d0 * inv, m1 = term * d1 * inv;
    float nc  = sqrtf(e0 * e0 + e1 * e1);
    float ivc = 1.0f / fmaxf(nc, 1e-12f);
    float c0 = termC * e0 * ivc, c1 = termC * e1 * ivc;
    int t = scene * NOBJ + j;
    out[t * 2 + 0] = m0 + c0;            out[t * 2 + 1] = m1 + c1;
    out[2 * NTOK + t * 2 + 0] = m0;      out[2 * NTOK + t * 2 + 1] = m1;
    out[4 * NTOK + t * 2 + 0] = c0;      out[4 * NTOK + t * 2 + 1] = c1;
  }
}

// ---------------------------------------------------------------------------
// Host launcher
// ---------------------------------------------------------------------------
extern "C" void kernel_launch(void* const* d_in, const int* in_sizes, int n_in,
                              void* d_out, int out_size, void* d_ws, size_t ws_size,
                              hipStream_t stream) {
  (void)in_sizes; (void)n_in; (void)out_size; (void)ws_size;
  const float* zdyn = (const float*)d_in[0];
  const float* zctx = (const float*)d_in[1];
  const float* zmc  = (const float*)d_in[2];
  const float* W_e  = (const float*)d_in[3];   const float* b_e  = (const float*)d_in[4];
  const float* W_i  = (const float*)d_in[5];   const float* b_i  = (const float*)d_in[6];
  const float* W_a  = (const float*)d_in[7];   const float* b_a  = (const float*)d_in[8];
  const float* W_2  = (const float*)d_in[9];   const float* b_2  = (const float*)d_in[10];
  const float* W_at = (const float*)d_in[11];  const float* b_at = (const float*)d_in[12];
  const float* W_at2 = (const float*)d_in[13]; const float* b_at2 = (const float*)d_in[14];
  const float* W_s1 = (const float*)d_in[15];  const float* b_s1 = (const float*)d_in[16];
  const float* W_s2 = (const float*)d_in[17];  const float* b_s2 = (const float*)d_in[18];
  const float* W_eC = (const float*)d_in[19];  const float* b_eC = (const float*)d_in[20];
  const float* W_iC = (const float*)d_in[21];  const float* b_iC = (const float*)d_in[22];
  const float* W_2C = (const float*)d_in[23];  const float* b_2C = (const float*)d_in[24];
  const float* W_s1C = (const float*)d_in[25]; const float* b_s1C = (const float*)d_in[26];
  const float* W_s2C = (const float*)d_in[27]; const float* b_s2C = (const float*)d_in[28];
  const float* g_M  = (const float*)d_in[29];  const float* be_M = (const float*)d_in[30];
  const float* g_C  = (const float*)d_in[31];  const float* be_C = (const float*)d_in[32];
  const float* W_d  = (const float*)d_in[33];  const float* W_dC = (const float*)d_in[34];

  char* ws = (char*)d_ws;
  _Float16* pwE  = (_Float16*)(ws + OFF_PWE);
  _Float16* pwI  = (_Float16*)(ws + OFF_PWI);
  _Float16* pwA  = (_Float16*)(ws + OFF_PWA);
  _Float16* pwEC = (_Float16*)(ws + OFF_PWEC);
  _Float16* pwIC = (_Float16*)(ws + OFF_PWIC);
  _Float16* pw2  = (_Float16*)(ws + OFF_PW2);
  _Float16* pw2C = (_Float16*)(ws + OFF_PW2C);
  _Float16* pwAT = (_Float16*)(ws + OFF_PWAT);
  float* wcM   = (float*)(ws + OFF_WCM);
  float* wcC   = (float*)(ws + OFF_WCC);
  float* s_ws  = (float*)(ws + OFF_S);
  float* sC_ws = (float*)(ws + OFF_SC);
  float* pm_ws = (float*)(ws + OFF_PM);
  float* part  = (float*)(ws + OFF_PART);
  float* stats = (float*)(ws + OFF_STATS);
  float* E_ws  = (float*)(ws + OFF_E);
  float* EC_ws = (float*)(ws + OFF_EC);
  float* out   = (float*)d_out;

  const int T_SMALL = 16 * 1 * 512;   // 8192
  const int T_BIG   = 16 * 8 * 512;   // 65536
  const int T_ATT   = 7 * 8 * 512;    // 28672
  k_pack<<<(T_SMALL + 255) / 256, 256, 0, stream>>>(W_e,  pwE,  1, 256, 28,  T_SMALL);
  k_pack<<<(T_SMALL + 255) / 256, 256, 0, stream>>>(W_i,  pwI,  1, 256, 30,  T_SMALL);
  k_pack<<<(T_SMALL + 255) / 256, 256, 0, stream>>>(W_a,  pwA,  1, 256, 28,  T_SMALL);
  k_pack<<<(T_SMALL + 255) / 256, 256, 0, stream>>>(W_eC, pwEC, 1, 256, 30,  T_SMALL);
  k_pack<<<(T_SMALL + 255) / 256, 256, 0, stream>>>(W_iC, pwIC, 1, 256, 30,  T_SMALL);
  k_pack<<<(T_BIG   + 255) / 256, 256, 0, stream>>>(W_2,  pw2,  8, 256, 256, T_BIG);
  k_pack<<<(T_BIG   + 255) / 256, 256, 0, stream>>>(W_2C, pw2C, 8, 256, 256, T_BIG);
  k_pack<<<(T_ATT   + 255) / 256, 256, 0, stream>>>(W_at, pwAT, 8, 100, 256, T_ATT);

  k_comb<<<1, 256, 0, stream>>>(W_s1,  b_s1,  W_s2,  b_s2,  wcM);
  k_comb<<<1, 256, 0, stream>>>(W_s1C, b_s1C, W_s2C, b_s2C, wcC);

  k_scene<<<NSCENE, 256, 0, stream>>>(
      zdyn, zctx, zmc, b_e, b_i, b_a, b_eC, b_iC, b_2, b_2C, b_at, W_at2, b_at2,
      pwE, pwI, pwA, pwEC, pwIC, pw2, pw2C, pwAT, wcM, wcC,
      s_ws, sC_ws, pm_ws, part, E_ws, EC_ws);

  k_stats<<<1, 256, 0, stream>>>(part, g_M, be_M, g_C, be_C, stats);

  k_final<<<NSCENE, 256, 0, stream>>>(s_ws, sC_ws, pm_ws, E_ws, EC_ws, stats,
                                      W_d, W_dC, out);
}
